// Decoder_84731114815924
// MI455X (gfx1250) — matrix operand order repaired and checked by reading, not compiled, not run on hardware
//
#include <hip/hip_runtime.h>
#include <hip/hip_fp16.h>

typedef __attribute__((ext_vector_type(16))) _Float16 v16h;
typedef __attribute__((ext_vector_type(8)))  _Float16 v8h;
typedef __attribute__((ext_vector_type(8)))  float    v8f;

#define B_  4
#define N_  256
#define D_  32
#define H_  512
#define K2_ 256   // H/2

// ---------------- prep: f16 copies of z and W2^T -------------------------
__global__ void prep_small(const float* __restrict__ z, const float* __restrict__ W2,
                           _Float16* __restrict__ zh, _Float16* __restrict__ w2t) {
  int stride = gridDim.x * blockDim.x;
  int id = blockIdx.x * blockDim.x + threadIdx.x;
  for (int t = id; t < B_ * N_ * D_; t += stride) zh[t] = (_Float16)z[t];
  // w2t[k][h] = W2[h][k]  (so GEMM2 B-fragments are contiguous per lane)
  for (int t = id; t < K2_ * H_; t += stride) {
    int k = t / H_, h = t % H_;
    w2t[t] = (_Float16)W2[h * K2_ + k];
  }
}

// ---------------- prep: tmpT[bi][h][c] = sum_a z[bi][a] * W1[(a*D+c)][h] --
__global__ void prep_tmp(const float* __restrict__ z, const float* __restrict__ W1,
                         _Float16* __restrict__ tmpT) {
  int stride = gridDim.x * blockDim.x;
  int total = B_ * N_ * D_ * H_;          // 16.8M, fits in int
  for (int id = blockIdx.x * blockDim.x + threadIdx.x; id < total; id += stride) {
    int h  = id & (H_ - 1);               // h fastest -> coalesced W1 reads
    int r  = id >> 9;                     // / H_
    int c  = r & (D_ - 1);
    int bi = r >> 5;                      // / D_
    const float* zr = z + bi * D_;
    float acc = 0.f;
#pragma unroll
    for (int a = 0; a < D_; ++a) acc += zr[a] * W1[(a * D_ + c) * H_ + h];
    tmpT[(bi * H_ + h) * D_ + c] = (_Float16)acc;
  }
}

// ---------------- gfx1250 async DMA: global -> LDS (ASYNCcnt tracked) ----
__device__ __forceinline__ void async_b128(void* lds_dst, const void* gsrc) {
  // LDS aperture maps addr[31:0] -> LDS byte offset (ISA 10.2), so the low
  // 32 bits of the generic pointer are the dsaddr the async engine needs.
  unsigned l = (unsigned)(unsigned long long)(size_t)lds_dst;
  asm volatile("global_load_async_to_lds_b128 %0, %1, off"
               :: "v"(l), "v"(gsrc) : "memory");
}
__device__ __forceinline__ void wait_async() {
  asm volatile("s_wait_asynccnt 0x0" ::: "memory");
}

// ---------------- fragment gathers (CDNA5 WMMA f16 layouts) --------------
// A (16x32 f16): lane L<16 holds row L, K {0..7,16..23}; lane L+16: K {8..15,24..31}
__device__ __forceinline__ v16h loadA(const _Float16* s, int row0, int pitch,
                                      int koff, int lane) {
  int r = lane & 15, half = lane >> 4;
  const _Float16* p = s + (row0 + r) * pitch + koff + half * 8;
  v8h lo = *(const v8h*)p;
  v8h hi = *(const v8h*)(p + 16);
  v16h a;
#pragma unroll
  for (int e = 0; e < 8; ++e) { a[e] = lo[e]; a[e + 8] = hi[e]; }
  return a;
}
// B (32x16 f16) from [N][K] row-major: lane n<16 holds col n, K {0..7,16..23}; lane n+16: K {8..15,24..31}
__device__ __forceinline__ v16h loadB(const _Float16* s, int col0, int pitch,
                                      int koff, int lane) {
  int n = lane & 15, kh = lane >> 4;
  const _Float16* p = s + (col0 + n) * pitch + koff + kh * 8;
  v8h lo = *(const v8h*)p;
  v8h hi = *(const v8h*)(p + 16);
  v16h b;
#pragma unroll
  for (int e = 0; e < 8; ++e) { b[e] = lo[e]; b[e + 8] = hi[e]; }
  return b;
}

// ---------------- main fused kernel: one block per (b,i) -----------------
// LDS pitches (f16 elements), all multiples of 8 for 16B-aligned b128 access
#define ZP 40   // z rows   (256 x 32 used)
#define TP 40   // tmp rows (512 x 32 used)
#define WP 72   // w2 chunk (256 x 64 used)
#define HP 72   // h1 chunk (256 x 64 used)

__global__ __launch_bounds__(512, 1)
void contact_main(const _Float16* __restrict__ zh, const _Float16* __restrict__ tmpT,
                  const _Float16* __restrict__ w2t,
                  const float* __restrict__ b1, const float* __restrict__ b2,
                  const float* __restrict__ w3, const float* __restrict__ b3,
                  const float* __restrict__ motif, float* __restrict__ out) {
  extern __shared__ _Float16 smem[];
  _Float16* s_zh  = smem;                       // 256*ZP
  _Float16* s_tmp = s_zh  + 256 * ZP;           // 512*TP
  _Float16* s_w2  = s_tmp + 512 * TP;           // 256*WP
  _Float16* s_h1  = s_w2  + 256 * WP;           // 256*HP
  float*    s_lg  = (float*)(s_h1 + 256 * HP);  // 256 f32

  const int tid  = threadIdx.x;
  const int lane = tid & 31;
  const int w    = tid >> 5;          // wave id 0..15
  const int bi   = blockIdx.x;        // b*256 + i
  const int b    = bi >> 8;
  const int i    = bi & 255;

  // stage z[b] (256 rows x 32 f16) and tmpT[b,i] (512 rows x 32 f16) via async DMA
  if (tid < 256) {
#pragma unroll
    for (int e = 0; e < 4; ++e)
      async_b128(s_zh + tid * ZP + e * 8, zh + (b * N_ + tid) * D_ + e * 8);
  }
#pragma unroll
  for (int e = 0; e < 4; ++e)
    async_b128(s_tmp + tid * TP + e * 8, tmpT + (bi * H_ + tid) * D_ + e * 8);
  if (tid < 256) s_lg[tid] = 0.f;
  wait_async();                       // drain own ASYNCcnt before the first barrier

  // GEMM2 accumulators: wave owns 64(j) x 64(k) region = 4x4 WMMA tiles
  v8f acc[4][4];
#pragma unroll
  for (int mt = 0; mt < 4; ++mt)
#pragma unroll
    for (int nt = 0; nt < 4; ++nt) acc[mt][nt] = v8f{};

  const int wj = (w & 3) * 64;   // j block
  const int wk = (w >> 2) * 64;  // k block
  const int kRow  = tid >> 1;    // staging row for W2 chunk
  const int kPart = tid & 1;

  for (int hc = 0; hc < 8; ++hc) {         // 8 chunks of 64 over H
    __syncthreads();                        // prev GEMM2 done -> reuse s_w2/s_h1
    // async-stage W2^T chunk (256 k-rows x 64 h-slice) while GEMM1 runs below
    {
      const _Float16* src = w2t + kRow * H_ + hc * 64 + kPart * 32;
      _Float16* dst = s_w2 + kRow * WP + kPart * 32;
      async_b128(dst,      src);
      async_b128(dst + 8,  src + 8);
      async_b128(dst + 16, src + 16);
      async_b128(dst + 24, src + 24);
      if (hc < 7)                           // pre-warm next chunk into caches
        __builtin_prefetch(src + 64, 0, 0);
    }
    // GEMM1: h1[j, hc*64 .. +64] = relu(z[b] @ tmpT + b1); wave w -> j-strip w*16
    {
      v16h aZ = loadA(s_zh, w * 16, ZP, 0, lane);
#pragma unroll
      for (int ht = 0; ht < 4; ++ht) {
        v16h bT = loadB(s_tmp, hc * 64 + ht * 16, TP, 0, lane);
        v8f c = v8f{};
        c = __builtin_amdgcn_wmma_f32_16x16x32_f16(false, aZ, false, bT,
                                                   (short)0, c, false, false);
        const int hcol = hc * 64 + ht * 16 + (lane & 15);
        const float bias = b1[hcol];
        const int rowbase = w * 16 + ((lane >> 4) ? 8 : 0);
        const int hl = ht * 16 + (lane & 15);
#pragma unroll
        for (int r2 = 0; r2 < 8; ++r2) {
          float v = c[r2] + bias;
          v = v > 0.f ? v : 0.f;
          s_h1[(rowbase + r2) * HP + hl] = (_Float16)v;
        }
      }
    }
    wait_async();                           // W2 chunk landed in LDS
    __syncthreads();                        // h1 chunk + w2 chunk ready
    // GEMM2: acc += h1_chunk @ W2_chunk   (K = 64 -> two K=32 WMMA steps)
#pragma unroll
    for (int ks = 0; ks < 2; ++ks) {
      v16h aF[4];
#pragma unroll
      for (int mt = 0; mt < 4; ++mt) aF[mt] = loadA(s_h1, wj + mt * 16, HP, ks * 32, lane);
#pragma unroll
      for (int nt = 0; nt < 4; ++nt) {
        v16h bF = loadB(s_w2, wk + nt * 16, WP, ks * 32, lane);
#pragma unroll
        for (int mt = 0; mt < 4; ++mt)
          acc[mt][nt] = __builtin_amdgcn_wmma_f32_16x16x32_f16(
              false, aF[mt], false, bF, (short)0, acc[mt][nt], false, false);
      }
    }
  }

  // GEMM3 epilogue: logits[j] += sum_k relu(acc + b2[k]) * w3[k]
#pragma unroll
  for (int mt = 0; mt < 4; ++mt) {
    float psum[8];
#pragma unroll
    for (int r2 = 0; r2 < 8; ++r2) psum[r2] = 0.f;
#pragma unroll
    for (int nt = 0; nt < 4; ++nt) {
      const int kcol = wk + nt * 16 + (lane & 15);
      const float bb = b2[kcol];
      const float ww = w3[kcol];
#pragma unroll
      for (int r2 = 0; r2 < 8; ++r2) {
        float v = acc[mt][nt][r2] + bb;
        v = v > 0.f ? v : 0.f;
        psum[r2] += v * ww;
      }
    }
#pragma unroll
    for (int r2 = 0; r2 < 8; ++r2) {
      float v = psum[r2];
#pragma unroll
      for (int off = 8; off >= 1; off >>= 1) v += __shfl_xor(v, off, 16);
      if ((lane & 15) == 0) {
        const int row = wj + mt * 16 + ((lane >> 4) ? 8 : 0) + r2;
        atomicAdd(&s_lg[row], v);           // ds_add_f32
      }
    }
  }
  __syncthreads();

  // masked sigmoid + write (out[0..BNN) = contact_map, out[BNN..2BNN) = logits)
  if (tid < 256) {
    const float lg = s_lg[tid] + b3[0];
    const float m  = motif[b * N_ + i] * motif[b * N_ + tid];
    const float lgm = lg * m;
    out[B_ * N_ * N_ + bi * N_ + tid] = lgm;
    out[bi * N_ + tid] = 1.f / (1.f + __expf(-lgm));
  }
}

// -------------------------------------------------------------------------
extern "C" void kernel_launch(void* const* d_in, const int* in_sizes, int n_in,
                              void* d_out, int out_size, void* d_ws, size_t ws_size,
                              hipStream_t stream) {
  const float* z     = (const float*)d_in[0];
  const float* motif = (const float*)d_in[1];
  // d_in[2] = residue_mask (unused by reference outputs)
  const float* W1 = (const float*)d_in[3];
  const float* b1 = (const float*)d_in[4];
  const float* W2 = (const float*)d_in[5];
  const float* b2 = (const float*)d_in[6];
  const float* W3 = (const float*)d_in[7];
  const float* b3 = (const float*)d_in[8];
  float* out = (float*)d_out;

  _Float16* zh   = (_Float16*)d_ws;            // B*N*D       = 32768 f16
  _Float16* w2t  = zh + B_ * N_ * D_;          // K2*H        = 131072 f16
  _Float16* tmpT = w2t + K2_ * H_;             // B*N*H*D     = 16.8M f16 (~32MB)

  prep_small<<<256, 256, 0, stream>>>(z, W2, zh, w2t);
  prep_tmp<<<2048, 256, 0, stream>>>(z, W1, tmpT);

  size_t lds = (size_t)(256 * ZP + 512 * TP + 256 * WP + 256 * HP) * sizeof(_Float16)
             + 256 * sizeof(float);
  contact_main<<<B_ * N_, 512, lds, stream>>>(zh, tmpT, w2t, b1, b2, W3, b3, motif, out);
}
